// TransformerModel_46213848104974
// MI455X (gfx1250) — matrix operand order repaired and checked by reading, not compiled, not run on hardware
//
#include <hip/hip_runtime.h>

// ---------------------------------------------------------------------------
// MI455X (gfx1250) transformer forward.
// All GEMMs via v_wmma_f32_16x16x32_bf16 (wave32). Block tile 128x128,
// wave tile 32x64 (8 WMMAs / K-step / wave), double-buffered LDS staging.
// Problem sizes: B=4, T=2048, D=1024, V=32000, BT=8192.
// ---------------------------------------------------------------------------

typedef __attribute__((ext_vector_type(16))) __bf16 v16bf;
typedef __attribute__((ext_vector_type(8)))  float  v8f;

struct __attribute__((may_alias)) U4 { unsigned x, y, z, w; };
union FragAB { U4 u[2]; v16bf v; };

__device__ __forceinline__ unsigned short f2bf(float f) {
    unsigned u = __builtin_bit_cast(unsigned, f);
    unsigned r = u + 0x7FFFu + ((u >> 16) & 1u);   // round-to-nearest-even
    return (unsigned short)(r >> 16);
}

// ---------------------------------------------------------------------------
// 1) Embedding: h[bt,d] = tok_emb[x[bt],d] + pos_emb[t,d]  ->  bf16
// ---------------------------------------------------------------------------
__global__ __launch_bounds__(256)
void embed_kernel(const int* __restrict__ x,
                  const float* __restrict__ tok_emb,
                  const float* __restrict__ pos_emb,
                  unsigned short* __restrict__ h)
{
    const int D = 1024;
    size_t idx = (size_t)blockIdx.x * blockDim.x + threadIdx.x;   // < 8192*1024
    int bt = (int)(idx >> 10);
    int d  = (int)(idx & 1023);
    int t  = bt & 2047;
    int tok = x[bt];
    float v = tok_emb[(size_t)tok * D + d] + pos_emb[(size_t)t * D + d];
    h[idx] = f2bf(v);
}

// ---------------------------------------------------------------------------
// 2) Weight convert+transpose: W [K,N] fp32 -> Wt [N,K] bf16 (LDS-tiled)
// ---------------------------------------------------------------------------
__global__ __launch_bounds__(256)
void convT_kernel(const float* __restrict__ W, unsigned short* __restrict__ Wt,
                  int K, int N)
{
    __shared__ float tile[32][33];
    int kBase = blockIdx.y * 32;
    int nBase = blockIdx.x * 32;
    for (int i = threadIdx.y; i < 32; i += 8)
        tile[i][threadIdx.x] = W[(size_t)(kBase + i) * N + nBase + threadIdx.x];
    __syncthreads();
    for (int i = threadIdx.y; i < 32; i += 8)
        Wt[(size_t)(nBase + i) * K + kBase + threadIdx.x] = f2bf(tile[threadIdx.x][i]);
}

// ---------------------------------------------------------------------------
// 3) Tiled WMMA GEMM:  C = scale * (A @ Bt^T) + bias  (+ optional causal mask)
//    A  : [M,K] bf16 row-major            (per-batch stride sA, elements)
//    Bt : [N,K] bf16 row-major (= B^T)    (per-batch stride sB)
//    C  : MODE 0 -> f32 [M,ldC]; MODE 1 -> bf16 [M,ldC];
//         MODE 2 -> bf16 transposed, C[n*ldC + m]        (stride sC, elements)
//    Block 128x128, 8 waves as 4(M)x2(N); wave = 32x64 = 2x4 16x16 tiles.
//    Double-buffered LDS (40 KB), one barrier per 32-K step.
// ---------------------------------------------------------------------------
#define BM  128
#define BN  128
#define TK  32
#define LDT 40                  // padded LDS row stride (halfs): 80B, bank-spread
#define TBUF (BM * LDT)         // halfs per tile buffer

template <int MODE>
__global__ __launch_bounds__(256)
void gemm_bf16_wmma(const unsigned short* __restrict__ A,
                    const unsigned short* __restrict__ Bt,
                    const float* __restrict__ bias,
                    void* __restrict__ Cout,
                    int M, int N, int K,
                    long long sA, long long sB, long long sC,
                    float scale, int causal, int ldC)
{
    __shared__ unsigned short As[2 * TBUF];
    __shared__ unsigned short Bs[2 * TBUF];

    const int b = blockIdx.z;
    A  += (long long)b * sA;
    Bt += (long long)b * sB;

    const int m0 = blockIdx.y * BM;
    const int n0 = blockIdx.x * BN;

    const int tid    = threadIdx.x;
    const int lane   = tid & 31;
    const int wave   = tid >> 5;     // 0..7
    const int wm     = wave >> 1;    // 0..3 : 32-row strip
    const int wn     = wave & 1;     // 0..1 : 64-col strip
    const int half16 = lane >> 4;    // lane group 0/1
    const int l15    = lane & 15;

    // Staging: 256 threads; each moves 2x16B of A and of B per K-step.
    const int ldRow = tid >> 1;          // 0..127
    const int c0    = (tid & 1) * 8;     // half chunks {0,16} / {8,24}

    // Fragment K offsets per ISA 7.12.2 (wave32):
    //  A (16x32): lanes0-15 K={0..7,16..23}, lanes16-31 K={8..15,24..31}
    //  B (32x16): lanes0-15 K={0..7,16..23}, lanes16-31 K={8..15,24..31}
    const int kbA = half16 * 8;
    const int kbB = half16 * 8;

    v8f c[2][4];
    #pragma unroll
    for (int i = 0; i < 2; ++i)
        #pragma unroll
        for (int j = 0; j < 4; ++j) c[i][j] = (v8f){};

    // Blocks entirely above the causal diagonal produce only -inf: skip math.
    const bool fullMasked = causal && (n0 > m0 + (BM - 1));

    if (!fullMasked) {
        const size_t aRow = (size_t)(m0 + ldRow) * K;
        const size_t bRow = (size_t)(n0 + ldRow) * K;

        // preload K-tile 0 into registers
        U4 ra0 = *(const U4*)&A [aRow + c0];
        U4 ra1 = *(const U4*)&A [aRow + c0 + 16];
        U4 rb0 = *(const U4*)&Bt[bRow + c0];
        U4 rb1 = *(const U4*)&Bt[bRow + c0 + 16];

        int cur = 0;
        for (int kk = 0; kk < K; kk += TK) {
            unsigned short* as = As + cur * TBUF;
            unsigned short* bs = Bs + cur * TBUF;
            *(U4*)&as[ldRow * LDT + c0]      = ra0;
            *(U4*)&as[ldRow * LDT + c0 + 16] = ra1;
            *(U4*)&bs[ldRow * LDT + c0]      = rb0;
            *(U4*)&bs[ldRow * LDT + c0 + 16] = rb1;
            __syncthreads();

            const int kn = kk + TK;
            if (kn < K) {   // stage next tile while WMMAs run
                ra0 = *(const U4*)&A [aRow + kn + c0];
                ra1 = *(const U4*)&A [aRow + kn + c0 + 16];
                rb0 = *(const U4*)&Bt[bRow + kn + c0];
                rb1 = *(const U4*)&Bt[bRow + kn + c0 + 16];
                if (kn + TK < K) {          // global_prefetch_b8 tile after next
                    __builtin_prefetch(&A [aRow + kn + TK + c0], 0, 1);
                    __builtin_prefetch(&Bt[bRow + kn + TK + c0], 0, 1);
                }
            }

            FragAB a[2], bf[4];
            #pragma unroll
            for (int i = 0; i < 2; ++i) {
                const int ar = wm * 32 + i * 16 + l15;
                a[i].u[0] = *(const U4*)&as[ar * LDT + kbA];
                a[i].u[1] = *(const U4*)&as[ar * LDT + kbA + 16];
            }
            #pragma unroll
            for (int j = 0; j < 4; ++j) {
                const int br = wn * 64 + j * 16 + l15;
                bf[j].u[0] = *(const U4*)&bs[br * LDT + kbB];
                bf[j].u[1] = *(const U4*)&bs[br * LDT + kbB + 16];
            }
            #pragma unroll
            for (int i = 0; i < 2; ++i)
                #pragma unroll
                for (int j = 0; j < 4; ++j)
                    c[i][j] = __builtin_amdgcn_wmma_f32_16x16x32_bf16(
                        false, a[i].v, false, bf[j].v, (short)0, c[i][j],
                        false, false);
            cur ^= 1;
        }
    }

    // Epilogue. C/D layout: VGPR r -> M = r + 8*half16 ; N = lane&15.
    float*          Cf = (float*)Cout          + (long long)b * sC;
    unsigned short* Ch = (unsigned short*)Cout + (long long)b * sC;

    #pragma unroll
    for (int i = 0; i < 2; ++i) {
        const int rowBase = m0 + wm * 32 + i * 16 + half16 * 8;
        #pragma unroll
        for (int j = 0; j < 4; ++j) {
            const int col = n0 + wn * 64 + j * 16 + l15;
            const float bv = bias ? bias[col] : 0.0f;
            #pragma unroll
            for (int r = 0; r < 8; ++r) {
                const int row = rowBase + r;
                float v = c[i][j][r] * scale + bv;
                if (causal && col > row) v = -__builtin_inff();
                if (MODE == 0)      Cf[(size_t)row * ldC + col] = v;
                else if (MODE == 1) Ch[(size_t)row * ldC + col] = f2bf(v);
                else                Ch[(size_t)col * ldC + row] = f2bf(v);
            }
        }
    }
}

// ---------------------------------------------------------------------------
// 4) Row softmax (2048 cols, -inf masked entries -> 0) -> bf16 probs
// ---------------------------------------------------------------------------
__global__ __launch_bounds__(256)
void softmax_kernel(const float* __restrict__ S, unsigned short* __restrict__ P,
                    int cols)
{
    __shared__ float red[256];
    const size_t base = (size_t)blockIdx.x * cols;
    const float* s = S + base;
    unsigned short* p = P + base;
    const int tid = threadIdx.x;

    float mx = -__builtin_inff();
    for (int c = tid; c < cols; c += 256) mx = fmaxf(mx, s[c]);
    red[tid] = mx; __syncthreads();
    for (int off = 128; off; off >>= 1) {
        if (tid < off) red[tid] = fmaxf(red[tid], red[tid + off]);
        __syncthreads();
    }
    mx = red[0]; __syncthreads();

    float sum = 0.0f;
    for (int c = tid; c < cols; c += 256) sum += __expf(s[c] - mx);
    red[tid] = sum; __syncthreads();
    for (int off = 128; off; off >>= 1) {
        if (tid < off) red[tid] += red[tid + off];
        __syncthreads();
    }
    const float inv = 1.0f / red[0];
    for (int c = tid; c < cols; c += 256) p[c] = f2bf(__expf(s[c] - mx) * inv);
}

// ---------------------------------------------------------------------------
// 5) Per-row cross-entropy from logits (32000 cols), then deterministic reduce
// ---------------------------------------------------------------------------
__global__ __launch_bounds__(256)
void row_loss_kernel(const float* __restrict__ logits,
                     const int* __restrict__ target,
                     float* __restrict__ rl)
{
    __shared__ float red[256];
    const int cols = 32000;
    const size_t base = (size_t)blockIdx.x * cols;
    const float* lr = logits + base;
    const int tid = threadIdx.x;

    float mx = -__builtin_inff();
    for (int c = tid; c < cols; c += 256) mx = fmaxf(mx, lr[c]);
    red[tid] = mx; __syncthreads();
    for (int off = 128; off; off >>= 1) {
        if (tid < off) red[tid] = fmaxf(red[tid], red[tid + off]);
        __syncthreads();
    }
    mx = red[0]; __syncthreads();

    float sum = 0.0f;
    for (int c = tid; c < cols; c += 256) sum += __expf(lr[c] - mx);
    red[tid] = sum; __syncthreads();
    for (int off = 128; off; off >>= 1) {
        if (tid < off) red[tid] += red[tid + off];
        __syncthreads();
    }
    if (tid == 0) {
        const int tgt = target[blockIdx.x];
        rl[blockIdx.x] = -(lr[tgt] - mx - __logf(red[0]));
    }
}

__global__ __launch_bounds__(256)
void reduce_loss_kernel(const float* __restrict__ rl, float* __restrict__ out, int n)
{
    __shared__ float red[256];
    const int tid = threadIdx.x;
    float s = 0.0f;
    for (int i = tid; i < n; i += 256) s += rl[i];
    red[tid] = s; __syncthreads();
    for (int off = 128; off; off >>= 1) {
        if (tid < off) red[tid] += red[tid + off];
        __syncthreads();
    }
    if (tid == 0) out[0] = red[0] / (float)n;
}

// ---------------------------------------------------------------------------
// Host-side orchestration
// ---------------------------------------------------------------------------
extern "C" void kernel_launch(void* const* d_in, const int* in_sizes, int n_in,
                              void* d_out, int out_size, void* d_ws, size_t ws_size,
                              hipStream_t stream)
{
    (void)in_sizes; (void)n_in; (void)out_size; (void)ws_size;
    const int B = 4, T = 2048, D = 1024, V = 32000;
    const int BT = B * T;                       // 8192

    const int*   x       = (const int*)  d_in[0];
    const int*   target  = (const int*)  d_in[1];
    const float* tok_emb = (const float*)d_in[2];
    const float* pos_emb = (const float*)d_in[3];
    const float* Wq      = (const float*)d_in[4];
    const float* bq      = (const float*)d_in[5];
    const float* Wk      = (const float*)d_in[6];
    const float* bk      = (const float*)d_in[7];
    const float* Wv      = (const float*)d_in[8];
    const float* bv      = (const float*)d_in[9];
    const float* Wlm     = (const float*)d_in[10];
    const float* blm     = (const float*)d_in[11];

    float* logits = (float*)d_out;                         // [8192, 32000]
    float* loss   = (float*)d_out + (size_t)BT * V;        // [1]

    // workspace carve (256B aligned)
    char* w = (char*)d_ws;
    size_t off = 0;
    auto carve = [&](size_t bytes) {
        char* p = w + off;
        off += (bytes + 255) & ~(size_t)255;
        return p;
    };
    unsigned short* h_bf   = (unsigned short*)carve((size_t)BT * D * 2);
    unsigned short* q_bf   = (unsigned short*)carve((size_t)BT * D * 2);
    unsigned short* k_bf   = (unsigned short*)carve((size_t)BT * D * 2);
    unsigned short* vT_bf  = (unsigned short*)carve((size_t)B * D * T * 2);  // [b][d][s]
    unsigned short* y_bf   = (unsigned short*)carve((size_t)BT * D * 2);
    unsigned short* WqT    = (unsigned short*)carve((size_t)D * D * 2);
    unsigned short* WkT    = (unsigned short*)carve((size_t)D * D * 2);
    unsigned short* WvT    = (unsigned short*)carve((size_t)D * D * 2);
    unsigned short* WlmT   = (unsigned short*)carve((size_t)V * D * 2);
    float*          scores = (float*)         carve((size_t)B * T * T * 4);
    unsigned short* attn   = (unsigned short*)carve((size_t)B * T * T * 2);
    float*          rloss  = (float*)         carve((size_t)BT * 4);

    // 1) embed
    embed_kernel<<<(BT * D) / 256, 256, 0, stream>>>(x, tok_emb, pos_emb, h_bf);

    // 2) weight transposes -> bf16 [N,K]
    dim3 tb(32, 8);
    convT_kernel<<<dim3(D / 32, D / 32), tb, 0, stream>>>(Wq,  WqT,  D, D);
    convT_kernel<<<dim3(D / 32, D / 32), tb, 0, stream>>>(Wk,  WkT,  D, D);
    convT_kernel<<<dim3(D / 32, D / 32), tb, 0, stream>>>(Wv,  WvT,  D, D);
    convT_kernel<<<dim3(V / 32, D / 32), tb, 0, stream>>>(Wlm, WlmT, D, V);

    // 3) q = h@Wq + bq ; k = h@Wk + bk   (bf16 out, [8192,1024])
    gemm_bf16_wmma<1><<<dim3(D / BN, BT / BM, 1), 256, 0, stream>>>(
        h_bf, WqT, bq, q_bf, BT, D, D, 0, 0, 0, 1.0f, 0, D);
    gemm_bf16_wmma<1><<<dim3(D / BN, BT / BM, 1), 256, 0, stream>>>(
        h_bf, WkT, bk, k_bf, BT, D, D, 0, 0, 0, 1.0f, 0, D);

    // v = h@Wv + bv, stored transposed per batch: vT[b][d][s]  (B-operand layout)
    gemm_bf16_wmma<2><<<dim3(D / BN, T / BM, B), 256, 0, stream>>>(
        h_bf, WvT, bv, vT_bf, T, D, D,
        (long long)T * D, 0, (long long)D * T, 1.0f, 0, T);

    // 4) scores = (q @ k^T) / sqrt(D), causal-masked, f32 [b][t][s]
    gemm_bf16_wmma<0><<<dim3(T / BN, T / BM, B), 256, 0, stream>>>(
        q_bf, k_bf, nullptr, scores, T, T, D,
        (long long)T * D, (long long)T * D, (long long)T * T,
        0.03125f /* 1024^-0.5 */, 1, T);

    // 5) softmax rows -> bf16 probs
    softmax_kernel<<<BT, 256, 0, stream>>>(scores, attn, T);

    // 6) y = attn @ v   (Bt = vT), bf16 out [b][t][d]
    gemm_bf16_wmma<1><<<dim3(D / BN, T / BM, B), 256, 0, stream>>>(
        attn, vT_bf, nullptr, y_bf, T, D, T,
        (long long)T * T, (long long)D * T, (long long)T * D, 1.0f, 0, D);

    // 7) logits = y @ W_lm + b_lm  -> f32 straight into d_out
    gemm_bf16_wmma<0><<<dim3(V / BN, BT / BM, 1), 256, 0, stream>>>(
        y_bf, WlmT, blm, logits, BT, V, D, 0, 0, 0, 1.0f, 0, V);

    // 8) loss = -mean(logp[target])
    row_loss_kernel<<<BT, 256, 0, stream>>>(logits, target, rloss);
    reduce_loss_kernel<<<1, 256, 0, stream>>>(rloss, loss, BT);
}
